// GPT2Block_59296318489114
// MI455X (gfx1250) — matrix operand order repaired and checked by reading, not compiled, not run on hardware
//
#include <hip/hip_runtime.h>
#include <cstdint>
#include <cstddef>

// ---------------- problem sizes (GPT-2 block) ----------------
#define BB 4
#define SS 1024
#define EE 1024
#define HH 16
#define DD 64
#define FF 4096
#define MR (BB * SS) /* 4096 rows */

typedef __attribute__((ext_vector_type(16))) __bf16 v16bf;
typedef __attribute__((ext_vector_type(8)))  __bf16 bf16x8;
typedef __attribute__((ext_vector_type(8)))  float  v8f;
typedef __attribute__((ext_vector_type(4)))  unsigned v4u;
typedef __attribute__((ext_vector_type(8)))  int      v8i;
typedef __attribute__((ext_vector_type(4)))  int      v4i;

#if defined(__has_builtin)
#  if __has_builtin(__builtin_amdgcn_tensor_load_to_lds) && \
      __has_builtin(__builtin_amdgcn_s_wait_tensorcnt)
#    define HAVE_TDM 1
#  endif
#endif
#ifndef HAVE_TDM
#  define HAVE_TDM 0
#endif

__device__ __forceinline__ v16bf cat8(bf16x8 lo, bf16x8 hi) {
  return __builtin_shufflevector(lo, hi, 0, 1, 2, 3, 4, 5, 6, 7,
                                 8, 9, 10, 11, 12, 13, 14, 15);
}
__device__ __forceinline__ v8f wmma_bf16(v16bf a, v16bf b, v8f c) {
  // D = A(16x32 bf16) * B(32x16 bf16) + C(16x16 f32)
  return __builtin_amdgcn_wmma_f32_16x16x32_bf16(false, a, false, b,
                                                 (short)0, c, false, false);
}

// ---- CDNA5 async global->LDS copy (ASYNCcnt-tracked), inline asm ----
__device__ __forceinline__ void async_copy_b128(const void* gsrc, void* lds_dst) {
  const unsigned lofs = (unsigned)(uintptr_t)lds_dst;          // LDS byte offset
  const unsigned long long ga = (unsigned long long)(uintptr_t)gsrc;
  asm volatile("global_load_async_to_lds_b128 %0, %1, off"
               :: "v"(lofs), "v"(ga) : "memory");
}
__device__ __forceinline__ void wait_async_le4() {
  asm volatile("s_wait_asynccnt 0x4" ::: "memory");
}
__device__ __forceinline__ void wait_async_0() {
  asm volatile("s_wait_asynccnt 0x0" ::: "memory");
}

#if HAVE_TDM
// ---- CDNA5 Tensor Data Mover: 2-D tile (32 cols x 128 rows, bf16) -> LDS ----
// LDS padding: row = 32 elems = 64B = 16 DW (pad_interval=3), pad 8 DW
// (pad_amount=7) => LDS row stride 48 elems = 96B, matching fragment layout.
__device__ __forceinline__ void tdm_load_tile_128x32(const void* gsrc,
                                                     unsigned lds_off,
                                                     int K_elems) {
  const unsigned long long ga = (unsigned long long)(uintptr_t)gsrc;
  v4u g0;
  g0.x = 1u;                                       // count=1 (valid descriptor)
  g0.y = lds_off;                                  // lds_addr (bytes)
  g0.z = (unsigned)(ga & 0xFFFFFFFFu);             // global_addr[31:0]
  g0.w = (unsigned)((ga >> 32) & 0x01FFFFFFu)      // global_addr[56:32]
       | (2u << 30);                               // type=2 ("image")
  v8i g1;
  g1[0] = (1 << 16)                                // data_size: 2 bytes
        | (1 << 20)                                // pad_enable
        | (3 << 22)                                // pad_interval: 16 DWORDs
        | (7 << 25);                               // pad_amount: 8 DWORDs
  g1[1] = (K_elems & 0xFFFF) << 16;                // tensor_dim0[15:0]
  g1[2] = ((K_elems >> 16) & 0xFFFF)               // tensor_dim0[31:16]
        | (128 << 16);                             // tensor_dim1 = 128
  g1[3] = (32 << 16);                              // tile_dim0 = 32 elems
  g1[4] = 128;                                     // tile_dim1 = 128, tile_dim2=0
  g1[5] = K_elems;                                 // tensor_dim0_stride[31:0]
  g1[6] = 0;                                       // stride[47:32], dim1_stride lo
  g1[7] = 0;
  const v4i z4 = {0, 0, 0, 0};                     // groups 2/3 unused (2-D tile)
  const v8i z8 = {0, 0, 0, 0, 0, 0, 0, 0};
  __builtin_amdgcn_tensor_load_to_lds(g0, g1, z4, z4, z8, 0);
}
#endif

// ---------------- fp32 -> bf16 convert ----------------
__global__ __launch_bounds__(256) void cvt_bf16_kernel(const float* __restrict__ in,
                                                       __bf16* __restrict__ out, int n) {
  for (int i = blockIdx.x * 256 + threadIdx.x; i < n; i += gridDim.x * 256)
    out[i] = (__bf16)in[i];
}

// ---------------- LayerNorm: fp32 row -> bf16 row ----------------
__global__ __launch_bounds__(256) void ln_kernel(const float* __restrict__ in,
                                                 const float* __restrict__ sc,
                                                 const float* __restrict__ bi,
                                                 __bf16* __restrict__ out) {
  const int row = blockIdx.x;
  const int tid = threadIdx.x;
  const float* p = in + (size_t)row * EE;
  float v[4], s = 0.f, q = 0.f;
#pragma unroll
  for (int i = 0; i < 4; ++i) {
    v[i] = p[tid + i * 256];
    s += v[i];
    q += v[i] * v[i];
  }
  __shared__ float sbuf[256], qbuf[256];
  sbuf[tid] = s; qbuf[tid] = q;
  __syncthreads();
  for (int o = 128; o > 0; o >>= 1) {
    if (tid < o) { sbuf[tid] += sbuf[tid + o]; qbuf[tid] += qbuf[tid + o]; }
    __syncthreads();
  }
  const float mean = sbuf[0] * (1.f / EE);
  const float var  = qbuf[0] * (1.f / EE) - mean * mean;
  const float rstd = rsqrtf(var + 1e-5f);
#pragma unroll
  for (int i = 0; i < 4; ++i) {
    const int c = tid + i * 256;
    out[(size_t)row * EE + c] = (__bf16)((v[i] - mean) * rstd * sc[c] + bi[c]);
  }
}

// ---------------- WMMA GEMM: C[M,N] = A[M,K] * W[N,K]^T + bias ----------------
// block tile 128x128, 8 waves of 32x64, K-step 32. Double-buffered LDS
// (row stride 48 elems = 96B). Tiles arrive via the Tensor Data Mover
// (wave 0: A-tile, wave 1: W-tile, TENSORcnt-pipelined); falls back to
// global_load_async_to_lds_b128 if the TDM builtin is unavailable.
template <bool GELU, bool RESID>
__global__ __launch_bounds__(256) void gemm_bf16_kernel(
    const __bf16* __restrict__ A, const __bf16* __restrict__ W,
    const float* __restrict__ bias, const float* __restrict__ resid,
    float* __restrict__ out_f, __bf16* __restrict__ out_b,
    int M, int N, int K) {
  __shared__ __bf16 lds_a[2][128 * 48];
  __shared__ __bf16 lds_w[2][128 * 48];
  const int tid  = threadIdx.x;
  const int lane = tid & 31;
  const int wave = tid >> 5;
  const int wm = wave >> 1, wn = wave & 1;
  const int bm = blockIdx.y * 128, bn = blockIdx.x * 128;
  const int arow = lane & 15;
  const int k0a  = (lane >> 4) * 16;      // A-frag K half
  const int k0b  = (lane >> 4) * 16;      // B-frag K half
  v8f acc[2][4] = {};

#if HAVE_TDM
  auto issue_tile = [&](int buf, int k0) {
    if (wave == 0)
      tdm_load_tile_128x32(A + (size_t)bm * K + k0,
                           (unsigned)(uintptr_t)&lds_a[buf][0], K);
    else if (wave == 1)
      tdm_load_tile_128x32(W + (size_t)bn * K + k0,
                           (unsigned)(uintptr_t)&lds_w[buf][0], K);
  };
#else
  const int lrow = tid >> 1;              // 0..127
  const int lseg = (tid & 1) * 16;        // 0 / 16 (16 bf16 = 32B per thread)
  auto issue_tile = [&](int buf, int k0) {
    const __bf16* ag = A + (size_t)(bm + lrow) * K + k0 + lseg;
    const __bf16* wg = W + (size_t)(bn + lrow) * K + k0 + lseg;
    async_copy_b128(ag,     &lds_a[buf][lrow * 48 + lseg]);
    async_copy_b128(ag + 8, &lds_a[buf][lrow * 48 + lseg + 8]);
    async_copy_b128(wg,     &lds_w[buf][lrow * 48 + lseg]);
    async_copy_b128(wg + 8, &lds_w[buf][lrow * 48 + lseg + 8]);
  };
#endif

  const int KT = K >> 5;  // K / 32 tiles
  issue_tile(0, 0);
  for (int kt = 0; kt < KT; ++kt) {
    if (kt + 1 < KT) {
      issue_tile((kt + 1) & 1, (kt + 1) << 5);
#if HAVE_TDM
      if (wave < 2) __builtin_amdgcn_s_wait_tensorcnt(1);  // prev tile resident
#else
      wait_async_le4();
#endif
    } else {
#if HAVE_TDM
      if (wave < 2) __builtin_amdgcn_s_wait_tensorcnt(0);
#else
      wait_async_0();
#endif
    }
    __syncthreads();      // all copies visible to every wave

    const __bf16* la = lds_a[kt & 1];
    const __bf16* lw = lds_w[kt & 1];
    v16bf af[2], bfr[4];
#pragma unroll
    for (int mi = 0; mi < 2; ++mi) {
      const int r = wm * 32 + mi * 16 + arow;
      bf16x8 lo = *(const bf16x8*)&la[r * 48 + k0a];
      bf16x8 hi = *(const bf16x8*)&la[r * 48 + k0a + 8];
      af[mi] = cat8(lo, hi);
    }
#pragma unroll
    for (int ni = 0; ni < 4; ++ni) {
      const int c = wn * 64 + ni * 16 + arow;
      bfr[ni] = *(const v16bf*)&lw[c * 48 + k0b];
    }
#pragma unroll
    for (int mi = 0; mi < 2; ++mi)
#pragma unroll
      for (int ni = 0; ni < 4; ++ni)
        acc[mi][ni] = wmma_bf16(af[mi], bfr[ni], acc[mi][ni]);
    __syncthreads();      // reads done before next overwrite
  }

  const int rofs = (lane < 16) ? 0 : 8;
#pragma unroll
  for (int mi = 0; mi < 2; ++mi)
#pragma unroll
    for (int ni = 0; ni < 4; ++ni) {
      const int col = bn + wn * 64 + ni * 16 + arow;
      const float bb = bias[col];
#pragma unroll
      for (int j = 0; j < 8; ++j) {
        const int row = bm + wm * 32 + mi * 16 + j + rofs;
        float v = acc[mi][ni][j] + bb;
        if (GELU) {
          const float u = v;
          v = 0.5f * u * (1.f + tanhf(0.7978845608028654f * (u + 0.044715f * u * u * u)));
        }
        if (RESID)
          out_f[(size_t)row * N + col] = v + resid[(size_t)row * N + col];
        else
          out_b[(size_t)row * N + col] = (__bf16)v;
      }
    }
}

// ---------------- flash attention (no mask), bf16 WMMA ----------------
// grid: (S/128, B*H). 8 waves/block; wave owns 16 q rows; key blocks of 32.
// K tile arrives via async global->LDS copy; V is transposed through VGPRs.
__global__ __launch_bounds__(256) void attn_kernel(const __bf16* __restrict__ qkv,
                                                   __bf16* __restrict__ o) {
  __shared__ __bf16 Klds[32 * 80];        // [key][d], stride 80 elems = 160B
  __shared__ __bf16 Vt[64 * 48];          // [d][key], stride 48 elems = 96B
  __shared__ __bf16 Plds[8][16 * 40];     // per-wave P tile, stride 40 elems = 80B

  const int tid  = threadIdx.x;
  const int lane = tid & 31;
  const int wave = tid >> 5;
  const int bh = blockIdx.y;
  const int b = bh >> 4, h = bh & 15;
  const int qrow0   = blockIdx.x * 128 + wave * 16;
  const int rowbase = b * SS;
  const int qcol = h * DD;
  const int kcol = EE + h * DD;
  const int vcol = 2 * EE + h * DD;
  const int arow = lane & 15;
  const int k0a  = (lane >> 4) * 16;
  const int rofs = (lane < 16) ? 0 : 8;

  // Q A-frags (pre-scaled by 1/sqrt(D) = 0.125)
  v16bf qa[2];
#pragma unroll
  for (int dc = 0; dc < 2; ++dc) {
    const __bf16* src = qkv + (size_t)(rowbase + qrow0 + arow) * (3 * EE) + qcol + dc * 32;
    bf16x8 lo = *(const bf16x8*)(src + k0a);
    bf16x8 hi = *(const bf16x8*)(src + k0a + 8);
    v16bf v = cat8(lo, hi);
#pragma unroll
    for (int i = 0; i < 16; ++i) v[i] = (__bf16)((float)v[i] * 0.125f);
    qa[dc] = v;
  }

  v8f o_acc[4] = {};
  float mrow[8], lrow[8];
#pragma unroll
  for (int j = 0; j < 8; ++j) { mrow[j] = -3.0e38f; lrow[j] = 0.f; }

  for (int kb = 0; kb < SS; kb += 32) {
    {  // K tile: async copy; V tile: load + transpose through VGPRs
      const int key = tid >> 3;
      const int d0  = (tid & 7) * 8;
      const size_t grow = (size_t)(rowbase + kb + key) * (3 * EE);
      async_copy_b128(qkv + grow + kcol + d0, &Klds[key * 80 + d0]);
      bf16x8 vv = *(const bf16x8*)(qkv + grow + vcol + d0);
#pragma unroll
      for (int i = 0; i < 8; ++i) Vt[(d0 + i) * 48 + key] = vv[i];
    }
    wait_async_0();
    __syncthreads();

    // logits tile: S[16 x 32keys] = Q * K^T
    v8f sfr[2];
#pragma unroll
    for (int nc = 0; nc < 2; ++nc) {
      v8f c = {};
#pragma unroll
      for (int dc = 0; dc < 2; ++dc) {
        const int keyn = nc * 16 + arow;
        const int koff = dc * 32 + ((lane < 16) ? 0 : 16);
        v16bf bfr = *(const v16bf*)&Klds[keyn * 80 + koff];
        c = wmma_bf16(qa[dc], bfr, c);
      }
      sfr[nc] = c;
    }

    // flash softmax update (rows live in 16-lane halves)
    float alpha[8];
#pragma unroll
    for (int j = 0; j < 8; ++j) {
      float mx = fmaxf(sfr[0][j], sfr[1][j]);
      mx = fmaxf(mx, __shfl_xor(mx, 1, 32));
      mx = fmaxf(mx, __shfl_xor(mx, 2, 32));
      mx = fmaxf(mx, __shfl_xor(mx, 4, 32));
      mx = fmaxf(mx, __shfl_xor(mx, 8, 32));
      const float mnew = fmaxf(mrow[j], mx);
      const float a0 = __expf(mrow[j] - mnew);
      const float p0 = __expf(sfr[0][j] - mnew);
      const float p1 = __expf(sfr[1][j] - mnew);
      sfr[0][j] = p0; sfr[1][j] = p1;
      float rs = p0 + p1;
      rs += __shfl_xor(rs, 1, 32);
      rs += __shfl_xor(rs, 2, 32);
      rs += __shfl_xor(rs, 4, 32);
      rs += __shfl_xor(rs, 8, 32);
      lrow[j] = lrow[j] * a0 + rs;
      mrow[j] = mnew;
      alpha[j] = a0;
    }
#pragma unroll
    for (int dc = 0; dc < 4; ++dc)
#pragma unroll
      for (int j = 0; j < 8; ++j) o_acc[dc][j] *= alpha[j];

    // P: C-layout -> LDS -> A-layout
    {
      __bf16* P = Plds[wave];
#pragma unroll
      for (int nc = 0; nc < 2; ++nc)
#pragma unroll
        for (int j = 0; j < 8; ++j)
          P[(j + rofs) * 40 + nc * 16 + arow] = (__bf16)sfr[nc][j];
    }
    __syncthreads();
    v16bf pa;
    {
      const __bf16* P = Plds[wave];
      bf16x8 lo = *(const bf16x8*)&P[arow * 40 + k0a];
      bf16x8 hi = *(const bf16x8*)&P[arow * 40 + k0a + 8];
      pa = cat8(lo, hi);
    }
    // O += P * V
#pragma unroll
    for (int dc = 0; dc < 4; ++dc) {
      const int dn = dc * 16 + arow;
      const int koff = (lane < 16) ? 0 : 16;
      v16bf bfr = *(const v16bf*)&Vt[dn * 48 + koff];
      o_acc[dc] = wmma_bf16(pa, bfr, o_acc[dc]);
    }
    __syncthreads();
  }

  // normalize + store bf16 [B*S, E]
#pragma unroll
  for (int dc = 0; dc < 4; ++dc)
#pragma unroll
    for (int j = 0; j < 8; ++j) {
      const float v = o_acc[dc][j] / lrow[j];
      const int r = qrow0 + j + rofs;
      const int c = h * DD + dc * 16 + arow;
      o[(size_t)(rowbase + r) * EE + c] = (__bf16)v;
    }
}

// ---------------- host orchestration ----------------
extern "C" void kernel_launch(void* const* d_in, const int* in_sizes, int n_in,
                              void* d_out, int out_size, void* d_ws, size_t ws_size,
                              hipStream_t stream) {
  const float* x         = (const float*)d_in[0];
  const float* ln1_scale = (const float*)d_in[1];
  const float* ln1_bias  = (const float*)d_in[2];
  const float* w_attn    = (const float*)d_in[3];
  const float* b_attn    = (const float*)d_in[4];
  const float* w_proj    = (const float*)d_in[5];
  const float* b_proj    = (const float*)d_in[6];
  const float* ln2_scale = (const float*)d_in[7];
  const float* ln2_bias  = (const float*)d_in[8];
  const float* w_fc      = (const float*)d_in[9];
  const float* b_fc      = (const float*)d_in[10];
  const float* w_proj2   = (const float*)d_in[11];
  const float* b_proj2   = (const float*)d_in[12];
  float* out = (float*)d_out;

  char* base = (char*)d_ws;
  size_t off = 0;
  auto alloc = [&](size_t bytes) -> char* {
    char* p = base + off;
    off = (off + bytes + 255) & ~(size_t)255;
    return p;
  };
  __bf16* wb_attn  = (__bf16*)alloc((size_t)3 * EE * EE * 2);
  __bf16* wb_proj  = (__bf16*)alloc((size_t)EE * EE * 2);
  __bf16* wb_fc    = (__bf16*)alloc((size_t)FF * EE * 2);
  __bf16* wb_proj2 = (__bf16*)alloc((size_t)EE * FF * 2);
  __bf16* h_ln     = (__bf16*)alloc((size_t)MR * EE * 2);
  __bf16* qkv      = (__bf16*)alloc((size_t)MR * 3 * EE * 2);
  __bf16* attn_o   = (__bf16*)alloc((size_t)MR * EE * 2);
  float*  x1       = (float*)alloc((size_t)MR * EE * 4);
  __bf16* fc_out   = (__bf16*)alloc((size_t)MR * FF * 2);

  // weight conversion fp32 -> bf16
  cvt_bf16_kernel<<<2048, 256, 0, stream>>>(w_attn,  wb_attn,  3 * EE * EE);
  cvt_bf16_kernel<<<2048, 256, 0, stream>>>(w_proj,  wb_proj,  EE * EE);
  cvt_bf16_kernel<<<2048, 256, 0, stream>>>(w_fc,    wb_fc,    FF * EE);
  cvt_bf16_kernel<<<2048, 256, 0, stream>>>(w_proj2, wb_proj2, EE * FF);

  // LN1
  ln_kernel<<<MR, 256, 0, stream>>>(x, ln1_scale, ln1_bias, h_ln);
  // QKV: [4096,3072] = h_ln * w_attn^T + b_attn (bf16 out)
  gemm_bf16_kernel<false, false><<<dim3(3 * EE / 128, MR / 128), 256, 0, stream>>>(
      h_ln, wb_attn, b_attn, nullptr, nullptr, qkv, MR, 3 * EE, EE);
  // attention
  attn_kernel<<<dim3(SS / 128, BB * HH), 256, 0, stream>>>(qkv, attn_o);
  // proj + residual -> x1 (fp32)
  gemm_bf16_kernel<false, true><<<dim3(EE / 128, MR / 128), 256, 0, stream>>>(
      attn_o, wb_proj, b_proj, x, x1, nullptr, MR, EE, EE);
  // LN2 (reuse h_ln)
  ln_kernel<<<MR, 256, 0, stream>>>(x1, ln2_scale, ln2_bias, h_ln);
  // FC + gelu (bf16 out)
  gemm_bf16_kernel<true, false><<<dim3(FF / 128, MR / 128), 256, 0, stream>>>(
      h_ln, wb_fc, b_fc, nullptr, nullptr, fc_out, MR, FF, EE);
  // proj2 + residual -> d_out (fp32)
  gemm_bf16_kernel<false, true><<<dim3(EE / 128, MR / 128), 256, 0, stream>>>(
      fc_out, wb_proj2, b_proj2, x1, out, nullptr, MR, EE, FF);
}
